// Pointer_decoder_52974126629537
// MI455X (gfx1250) — compile-verified, matrix operand order checked by reading, not run on hardware
//
#include <hip/hip_runtime.h>
#include <hip/hip_bf16.h>

// Pointer-network decoder for MI455X (gfx1250).
//  - All GEMMs via v_wmma_f32_16x16x32_bf16 (bf16 inputs, f32 accum).
//  - enc_ref / enc_ref_g projections precomputed ONCE (2x2 register-blocked
//    WMMA GEMM) and stored bf16 so both (33.5 MB each) stay resident in the
//    192 MB L2 across all 127 decode steps.
//  - Small per-step GEMMs use a 1x1-tile WMMA GEMM (max grid parallelism).
//  - LSTM = single WMMA GEMM on concatenated [x|h] x [W_ih|W_hh]^T.
//  - Attention reductions are wave32 shuffle reductions; sampling is a
//    deterministic hash-Gumbel argmax (JAX threefry cannot be bit-matched).

#define B_ 512
#define S_ 128
#define H_ 256
#define BIGC 100000000.0f

typedef __attribute__((ext_vector_type(16))) __bf16 v16bf;
typedef __attribute__((ext_vector_type(8)))  float  v8f;

__device__ __forceinline__ v16bf load_frag16(const __bf16* p) {
  union { uint4 u[2]; v16bf v; } x;
  x.u[0] = *(const uint4*)(p);
  x.u[1] = *(const uint4*)(p + 8);
  return x.v;
}

__device__ __forceinline__ v8f wmma_bf16(v16bf a, v16bf b, v8f c) {
  return __builtin_amdgcn_wmma_f32_16x16x32_bf16(
      /*neg_a=*/false, a, /*neg_b=*/false, b,
      /*c_mod=*/(short)0, c, /*reuse_a=*/false, /*reuse_b=*/false);
}

__device__ __forceinline__ float sigm(float x) { return 1.0f / (1.0f + expf(-x)); }

// ---------------------------------------------------------------------------
// 1x1-tile WMMA GEMM: C[M x N] = A[M x K] * W[N x K]^T  (A,W bf16 row-major)
// block = 256 threads = 8 waves; wave tile 16x16; block tile 32(M) x 64(N)
// grid = (N/64, M/32). Used for the small per-step GEMMs (occupancy first).
// A-frag: lane r (0-15) holds row m_base+r, halves = 16 consecutive K.
// B-frag: lane r (0-15) holds col n_base+r, halves = 16 consecutive K
//         -> both frags are two contiguous b128 loads (W is (N,K) row-major).
// ---------------------------------------------------------------------------
template<bool OUT_BF16>
__global__ __launch_bounds__(256) void gemm_wmma(
    const __bf16* __restrict__ A, int lda,
    const __bf16* __restrict__ W, int ldw,
    float* __restrict__ Cf, __bf16* __restrict__ Cb, int ldc, int K) {
  const int wave = threadIdx.x >> 5;
  const int lane = threadIdx.x & 31;
  const int wm = wave & 1;
  const int wn = wave >> 1;
  const int m_base = blockIdx.y * 32 + wm * 16;
  const int n_base = blockIdx.x * 64 + wn * 16;
  const int hl = lane >> 4;
  const int r  = lane & 15;
  const __bf16* ap = A + (size_t)(m_base + r) * lda + hl * 16;
  const __bf16* bp = W + (size_t)(n_base + r) * ldw + hl * 16;
  v8f acc = {};
  for (int k0 = 0; k0 < K; k0 += 32) {
    v16bf af = load_frag16(ap + k0);
    v16bf bf = load_frag16(bp + k0);
    acc = wmma_bf16(af, bf, acc);
  }
  const int col = n_base + r;
#pragma unroll
  for (int i = 0; i < 8; ++i) {
    const int row = m_base + hl * 8 + i;
    if (OUT_BF16) Cb[(size_t)row * ldc + col] = (__bf16)acc[i];
    else          Cf[(size_t)row * ldc + col] = acc[i];
  }
}

// ---------------------------------------------------------------------------
// 2x2 register-blocked WMMA GEMM: wave tile 32x32 (4 accumulators), block =
// 8 waves arranged 2(M) x 4(N) -> block tile 64(M) x 128(N).
// grid = (N/128, M/64). Each A/B fragment is reused twice -> half the
// fragment load traffic and 4 independent WMMA chains for latency hiding.
// Used for the large one-time (B*S) x H projections.
// ---------------------------------------------------------------------------
__global__ __launch_bounds__(256) void gemm_wmma_2x2_bf16out(
    const __bf16* __restrict__ A, int lda,
    const __bf16* __restrict__ W, int ldw,
    __bf16* __restrict__ Cb, int ldc, int K) {
  const int wave = threadIdx.x >> 5;
  const int lane = threadIdx.x & 31;
  const int wm = wave & 1;
  const int wn = wave >> 1;
  const int m_base = blockIdx.y * 64 + wm * 32;
  const int n_base = blockIdx.x * 128 + wn * 32;
  const int hl = lane >> 4;
  const int r  = lane & 15;
  const __bf16* ap0 = A + (size_t)(m_base + r) * lda + hl * 16;
  const __bf16* ap1 = ap0 + (size_t)16 * lda;
  const __bf16* bp0 = W + (size_t)(n_base + r) * ldw + hl * 16;
  const __bf16* bp1 = bp0 + (size_t)16 * ldw;
  v8f acc00 = {}, acc01 = {}, acc10 = {}, acc11 = {};
  for (int k0 = 0; k0 < K; k0 += 32) {
    v16bf a0 = load_frag16(ap0 + k0);
    v16bf a1 = load_frag16(ap1 + k0);
    v16bf b0 = load_frag16(bp0 + k0);
    v16bf b1 = load_frag16(bp1 + k0);
    acc00 = wmma_bf16(a0, b0, acc00);
    acc01 = wmma_bf16(a0, b1, acc01);
    acc10 = wmma_bf16(a1, b0, acc10);
    acc11 = wmma_bf16(a1, b1, acc11);
  }
  const int col0 = n_base + r;
  const int col1 = col0 + 16;
#pragma unroll
  for (int i = 0; i < 8; ++i) {
    const int row0 = m_base + hl * 8 + i;
    const int row1 = row0 + 16;
    Cb[(size_t)row0 * ldc + col0] = (__bf16)acc00[i];
    Cb[(size_t)row0 * ldc + col1] = (__bf16)acc01[i];
    Cb[(size_t)row1 * ldc + col0] = (__bf16)acc10[i];
    Cb[(size_t)row1 * ldc + col1] = (__bf16)acc11[i];
  }
}

// ------------------------- prep / cast kernels -----------------------------
__global__ void k_cast_bf(const float* __restrict__ src, __bf16* __restrict__ dst, int n) {
  int i = blockIdx.x * blockDim.x + threadIdx.x;
  if (i < n) dst[i] = (__bf16)src[i];
}

// W_cat[n][k] = k<256 ? W_ih[n][k] : W_hh[n][k-256]   (1024 x 512)
__global__ void k_build_wcat(const float* __restrict__ wih, const float* __restrict__ whh,
                             __bf16* __restrict__ wcat) {
  int i = blockIdx.x * blockDim.x + threadIdx.x;
  if (i >= 1024 * 512) return;
  int n = i >> 9, k = i & 511;
  float v = (k < 256) ? wih[n * 256 + k] : whh[n * 256 + (k - 256)];
  wcat[i] = (__bf16)v;
}

__global__ void k_init_state(const float* __restrict__ enc, const float* __restrict__ h0,
                             const float* __restrict__ c0, float* __restrict__ h,
                             float* __restrict__ c, __bf16* __restrict__ Acat,
                             float* __restrict__ logp, float* __restrict__ out) {
  int idx = blockIdx.x * blockDim.x + threadIdx.x;   // B*H
  if (idx >= B_ * H_) return;
  int b = idx >> 8, j = idx & 255;
  float hv = h0[idx], cv = c0[idx];
  h[idx] = hv; c[idx] = cv;
  Acat[(size_t)b * 512 + 256 + j] = (__bf16)hv;
  Acat[(size_t)b * 512 + j] = (__bf16)enc[((size_t)b * S_ + (S_ - 1)) * H_ + j];
  if (j == 0) {
    logp[b] = 0.0f;
    out[(size_t)b * (S_ + 1) + 0]  = (float)(S_ - 1);   // depot leading
    out[(size_t)b * (S_ + 1) + S_] = (float)(S_ - 1);   // depot trailing
  }
}

__global__ void k_init_mask(float* __restrict__ mask) {
  int i = blockIdx.x * blockDim.x + threadIdx.x;   // B*S
  if (i < B_ * S_) mask[i] = ((i & (S_ - 1)) == (S_ - 1)) ? 1.0f : 0.0f;
}

// ------------------------------ LSTM cell ----------------------------------
__global__ void k_lstm(const float* __restrict__ gates, float* __restrict__ h,
                       float* __restrict__ c, __bf16* __restrict__ Acat) {
  int idx = blockIdx.x * blockDim.x + threadIdx.x;   // B*H
  if (idx >= B_ * H_) return;
  int b = idx >> 8, j = idx & 255;
  const float* gb = gates + (size_t)b * 1024;
  float ig = gb[j], fg = gb[256 + j], gg = gb[512 + j], og = gb[768 + j];
  float cc = sigm(fg) * c[idx] + sigm(ig) * tanhf(gg);
  float hh = sigm(og) * tanhf(cc);
  c[idx] = cc; h[idx] = hh;
  Acat[(size_t)b * 512 + 256 + j] = (__bf16)hh;
}

// out[b,s] = sum_h vvec[h] * tanh(encP[b,s,h] + add[b,h]); optional 10*tanh; -BIG*mask
__global__ __launch_bounds__(256) void k_att_rowsum(
    const __bf16* __restrict__ encP, const float* __restrict__ add,
    const float* __restrict__ vvec, const float* __restrict__ mask,
    float* __restrict__ out, int clip) {
  int wave = threadIdx.x >> 5, lane = threadIdx.x & 31;
  int row = blockIdx.x * 8 + wave;        // row = b*S + s
  int b = row >> 7;
  const __bf16* ep = encP + (size_t)row * H_;
  const float*  ad = add + (size_t)b * H_;
  float sum = 0.0f;
#pragma unroll
  for (int t = 0; t < 8; ++t) {
    int j = lane + 32 * t;
    sum += vvec[j] * tanhf((float)ep[j] + ad[j]);
  }
#pragma unroll
  for (int off = 16; off > 0; off >>= 1) sum += __shfl_xor(sum, off, 32);
  if (lane == 0) {
    if (clip) sum = 10.0f * tanhf(sum);
    out[row] = sum - BIGC * mask[row];
  }
}

// softmax over gs[b,:] then glimpse[b,o] = sum_s attn*enc_ref_g[b,s,o] + h[b,o]
__global__ __launch_bounds__(256) void k_softmax_glimpse(
    const float* __restrict__ gs, const __bf16* __restrict__ encg,
    const float* __restrict__ h, __bf16* __restrict__ glimpse_bf) {
  __shared__ float attn[S_];
  __shared__ float red[256];
  int b = blockIdx.x, tid = threadIdx.x;
  float val = (tid < S_) ? gs[(size_t)b * S_ + tid] : -3.4e38f;
  red[tid] = val; __syncthreads();
  for (int s = 128; s > 0; s >>= 1) { if (tid < s) red[tid] = fmaxf(red[tid], red[tid + s]); __syncthreads(); }
  float mx = red[0]; __syncthreads();
  float e = (tid < S_) ? expf(val - mx) : 0.0f;
  red[tid] = e; __syncthreads();
  for (int s = 128; s > 0; s >>= 1) { if (tid < s) red[tid] += red[tid + s]; __syncthreads(); }
  float inv = 1.0f / red[0];
  if (tid < S_) attn[tid] = e * inv;
  __syncthreads();
  float acc = h[(size_t)b * H_ + tid];
  const __bf16* ep = encg + (size_t)b * S_ * H_ + tid;
#pragma unroll 4
  for (int s = 0; s < S_; ++s) acc += attn[s] * (float)ep[(size_t)s * H_];
  glimpse_bf[(size_t)b * H_ + tid] = (__bf16)acc;
}

// ------------------------- sampling / bookkeeping --------------------------
__device__ __forceinline__ unsigned pcg(unsigned v) {
  unsigned s = v * 747796405u + 2891336453u;
  unsigned w = ((s >> ((s >> 28) + 4u)) ^ s) * 277803737u;
  return (w >> 22) ^ w;
}

__global__ __launch_bounds__(256) void k_sample(
    const float* __restrict__ scores, int step, float* __restrict__ mask,
    float* __restrict__ logp_sum, float* __restrict__ out,
    const float* __restrict__ enc, __bf16* __restrict__ Acat) {
  __shared__ float red[S_];
  __shared__ int   redi[S_];
  __shared__ int   pos_sh;
  __shared__ float lse_sh;
  int b = blockIdx.x, tid = threadIdx.x;
  float m = -3.4e38f, g = 0.0f;
  if (tid < S_) {
    m = scores[(size_t)b * S_ + tid];
    unsigned u = pcg(pcg((unsigned)step * 0x9E3779B9u + (unsigned)b) ^ pcg((unsigned)tid + 0x85EBCA6Bu));
    float uf = (float)(u >> 8) * (1.0f / 16777216.0f) + 1e-10f;
    g = -logf(-logf(uf));
  }
  // logsumexp(m)
  if (tid < S_) red[tid] = m;
  __syncthreads();
  for (int s = 64; s > 0; s >>= 1) { if (tid < s) red[tid] = fmaxf(red[tid], red[tid + s]); __syncthreads(); }
  float mx = red[0]; __syncthreads();
  if (tid < S_) red[tid] = expf(m - mx);
  __syncthreads();
  for (int s = 64; s > 0; s >>= 1) { if (tid < s) red[tid] += red[tid + s]; __syncthreads(); }
  if (tid == 0) lse_sh = mx + logf(red[0]);
  __syncthreads();
  // Gumbel-max argmax of m + g
  if (tid < S_) { red[tid] = m + g; redi[tid] = tid; }
  __syncthreads();
  for (int s = 64; s > 0; s >>= 1) {
    if (tid < s && red[tid + s] > red[tid]) { red[tid] = red[tid + s]; redi[tid] = redi[tid + s]; }
    __syncthreads();
  }
  if (tid == 0) {
    int pos = redi[0];
    pos_sh = pos;
    logp_sum[b] += scores[(size_t)b * S_ + pos] - lse_sh;
    mask[(size_t)b * S_ + pos] += 1.0f;
    out[(size_t)b * (S_ + 1) + 1 + step] = (float)pos;
  }
  __syncthreads();
  int pos = pos_sh;   // gather next input x = encoder_output[b, pos, :]
  Acat[(size_t)b * 512 + tid] = (__bf16)enc[((size_t)b * S_ + pos) * H_ + tid];
}

__global__ void k_write_logp(const float* __restrict__ logp, float* __restrict__ out) {
  int b = blockIdx.x * blockDim.x + threadIdx.x;
  if (b < B_) out[(size_t)B_ * (S_ + 1) + b] = logp[b];
}

// ---------------------------------------------------------------------------
extern "C" void kernel_launch(void* const* d_in, const int* in_sizes, int n_in,
                              void* d_out, int out_size, void* d_ws, size_t ws_size,
                              hipStream_t stream) {
  const float* enc   = (const float*)d_in[0];
  const float* h0    = (const float*)d_in[1];
  const float* c0    = (const float*)d_in[2];
  const float* W_ih  = (const float*)d_in[3];
  const float* W_hh  = (const float*)d_in[4];
  const float* W_ref = (const float*)d_in[5];
  const float* W_q   = (const float*)d_in[6];
  const float* v     = (const float*)d_in[7];
  const float* Wg_ref= (const float*)d_in[8];
  const float* Wg_q  = (const float*)d_in[9];
  const float* vg    = (const float*)d_in[10];
  float* out = (float*)d_out;

  char* p = (char*)d_ws;
  auto alloc = [&](size_t bytes) { char* r = p; p += (bytes + 255) & ~(size_t)255; return r; };

  const size_t NEL = (size_t)B_ * S_ * H_;                 // 16.7M
  __bf16* enc_bf     = (__bf16*)alloc(NEL * 2);            // 33.5 MB
  __bf16* encref_bf  = (__bf16*)alloc(NEL * 2);            // 33.5 MB (L2-resident)
  __bf16* encrefg_bf = (__bf16*)alloc(NEL * 2);            // 33.5 MB (L2-resident)
  __bf16* Wcat_bf    = (__bf16*)alloc((size_t)1024 * 512 * 2);
  __bf16* Wref_bf    = (__bf16*)alloc((size_t)H_ * H_ * 2);
  __bf16* Wq_bf      = (__bf16*)alloc((size_t)H_ * H_ * 2);
  __bf16* Wgref_bf   = (__bf16*)alloc((size_t)H_ * H_ * 2);
  __bf16* Wgq_bf     = (__bf16*)alloc((size_t)H_ * H_ * 2);
  __bf16* Acat       = (__bf16*)alloc((size_t)B_ * 512 * 2);   // [x | h] bf16
  float*  gates      = (float*)alloc((size_t)B_ * 1024 * 4);
  float*  h          = (float*)alloc((size_t)B_ * H_ * 4);
  float*  c          = (float*)alloc((size_t)B_ * H_ * 4);
  float*  hq         = (float*)alloc((size_t)B_ * H_ * 4);
  float*  q          = (float*)alloc((size_t)B_ * H_ * 4);
  float*  gs         = (float*)alloc((size_t)B_ * S_ * 4);
  float*  scores     = (float*)alloc((size_t)B_ * S_ * 4);
  float*  mask       = (float*)alloc((size_t)B_ * S_ * 4);
  __bf16* glimpse_bf = (__bf16*)alloc((size_t)B_ * H_ * 2);
  float*  logp       = (float*)alloc((size_t)B_ * 4);
  (void)ws_size; (void)in_sizes; (void)n_in; (void)out_size;

  // ---- one-time prep ----
  k_cast_bf<<<(int)((NEL + 255) / 256), 256, 0, stream>>>(enc, enc_bf, (int)NEL);
  k_build_wcat<<<(1024 * 512 + 255) / 256, 256, 0, stream>>>(W_ih, W_hh, Wcat_bf);
  k_cast_bf<<<(H_ * H_ + 255) / 256, 256, 0, stream>>>(W_ref,  Wref_bf,  H_ * H_);
  k_cast_bf<<<(H_ * H_ + 255) / 256, 256, 0, stream>>>(W_q,    Wq_bf,    H_ * H_);
  k_cast_bf<<<(H_ * H_ + 255) / 256, 256, 0, stream>>>(Wg_ref, Wgref_bf, H_ * H_);
  k_cast_bf<<<(H_ * H_ + 255) / 256, 256, 0, stream>>>(Wg_q,   Wgq_bf,   H_ * H_);
  k_init_state<<<(B_ * H_ + 255) / 256, 256, 0, stream>>>(enc, h0, c0, h, c, Acat, logp, out);
  k_init_mask<<<(B_ * S_ + 255) / 256, 256, 0, stream>>>(mask);

  // enc_ref / enc_ref_g projections: (B*S) x H GEMMs, 2x2 blocked, bf16 out
  gemm_wmma_2x2_bf16out<<<dim3(H_ / 128, (B_ * S_) / 64), 256, 0, stream>>>(
      enc_bf, H_, Wref_bf, H_, encref_bf, H_, H_);
  gemm_wmma_2x2_bf16out<<<dim3(H_ / 128, (B_ * S_) / 64), 256, 0, stream>>>(
      enc_bf, H_, Wgref_bf, H_, encrefg_bf, H_, H_);

  // ---- sequential decode: S-1 = 127 steps ----
  for (int step = 0; step < S_ - 1; ++step) {
    // gates = [x|h] @ [W_ih|W_hh]^T   (512 x 512 x 1024)
    gemm_wmma<false><<<dim3(1024 / 64, B_ / 32), 256, 0, stream>>>(
        Acat, 512, Wcat_bf, 512, gates, nullptr, 1024, 512);
    k_lstm<<<(B_ * H_ + 255) / 256, 256, 0, stream>>>(gates, h, c, Acat);

    // hq = h @ Wg_q^T
    gemm_wmma<false><<<dim3(H_ / 64, B_ / 32), 256, 0, stream>>>(
        Acat + 256, 512, Wgq_bf, H_, hq, nullptr, H_, H_);

    // gs = sum_h vg*tanh(enc_ref_g + hq) - BIG*mask
    k_att_rowsum<<<(B_ * S_) / 8, 256, 0, stream>>>(encrefg_bf, hq, vg, mask, gs, 0);
    k_softmax_glimpse<<<B_, 256, 0, stream>>>(gs, encrefg_bf, h, glimpse_bf);

    // q = glimpse @ W_q^T
    gemm_wmma<false><<<dim3(H_ / 64, B_ / 32), 256, 0, stream>>>(
        glimpse_bf, H_, Wq_bf, H_, q, nullptr, H_, H_);

    // scores = 10*tanh(sum_h v*tanh(enc_ref + q)) - BIG*mask
    k_att_rowsum<<<(B_ * S_) / 8, 256, 0, stream>>>(encref_bf, q, v, mask, scores, 1);

    // sample pos, accumulate logp, update mask, gather next x
    k_sample<<<B_, 256, 0, stream>>>(scores, step, mask, logp, out, enc, Acat);
  }

  k_write_logp<<<(B_ + 255) / 256, 256, 0, stream>>>(logp, out);
}
